// MaskedPointModel_20856361189605
// MI455X (gfx1250) — matrix operand order repaired and checked by reading, not compiled, not run on hardware
//
#include <hip/hip_runtime.h>

// ---------------------------------------------------------------------------
// Model dims
// ---------------------------------------------------------------------------
#define BB     32
#define NN     65536
#define NPATCH 64
#define KK     32
#define ED     128
#define DD     64
#define PD     32
#define DT     96      // DD + PD
#define FFD    2048
#define OUTC   11

// output packing (float elements)
#define OFF_R        0         // 32*64*32*11 = 720896
#define OFF_MASK     720896    // 2048
#define OFF_DEC      722944    // 196608
#define OFF_CENTERS  919552    // 6144
#define OFF_GIDX     925696    // 65536

// ---------------------------------------------------------------------------
// WMMA types / helpers (CDNA5 wave32, v_wmma_f32_16x16x32_f16)
// ---------------------------------------------------------------------------
typedef __attribute__((ext_vector_type(16))) _Float16 v16h;
typedef __attribute__((ext_vector_type(8)))  _Float16 v8h;
typedef __attribute__((ext_vector_type(8)))  float    v8f;

// A fragment (16x32 f16): lane l16 = row; two contiguous 8-half runs (16B each).
static __device__ inline v16h loadA16h(const _Float16* __restrict__ row, int kb, int half)
{
    v8h a = *(const v8h*)(row + kb + 8 * half);
    v8h b = *(const v8h*)(row + kb + 16 + 8 * half);
    return __builtin_shufflevector(a, b, 0, 1, 2, 3, 4, 5, 6, 7,
                                         8, 9, 10, 11, 12, 13, 14, 15);
}

// B fragment (32x16 f16): lane l16 = column (row of W); two contiguous 8-half runs (16B each).
static __device__ inline v16h loadB16h(const _Float16* __restrict__ row, int kb, int half)
{
    v8h a = *(const v8h*)(row + kb + 8 * half);
    v8h b = *(const v8h*)(row + kb + 16 + 8 * half);
    return __builtin_shufflevector(a, b, 0, 1, 2, 3, 4, 5, 6, 7,
                                         8, 9, 10, 11, 12, 13, 14, 15);
}

// ---------------------------------------------------------------------------
// GEMM: C[M,N] = act(A[M,K] @ W[N,K]^T + bias) (+ res)
// A,W pre-staged in f16. 32x32 macro-tile per wave (2x2 of 16x16).
// K%32==0, M%32==0 required. Ragged N via clamped W rows + store guards.
// Output to f32 (C) and/or f16 (Ch).
// ---------------------------------------------------------------------------
__global__ __launch_bounds__(256)
void gemm_wmma(const _Float16* __restrict__ A, const _Float16* __restrict__ W,
               const float* __restrict__ bias, const float* __restrict__ res,
               float* __restrict__ C, _Float16* __restrict__ Ch,
               int M, int N, int K, int act)
{
    int lane = threadIdx.x & 31;
    int wave = threadIdx.x >> 5;
    int tilesN = (N + 31) >> 5;
    int tilesM = M >> 5;
    int t = blockIdx.x * 8 + wave;            // uniform per wave -> EXEC all 1s
    if (t >= tilesM * tilesN) return;
    int tm = (t / tilesN) << 5;
    int tn = (t % tilesN) << 5;
    int half = lane >> 4;
    int l16  = lane & 15;

    const _Float16* Arow0 = A + (size_t)(tm + l16) * K;
    const _Float16* Arow1 = A + (size_t)(tm + 16 + l16) * K;
    int wr0 = tn + l16, wr1 = tn + 16 + l16;
    const _Float16* Wrow0 = W + (size_t)(wr0 < N ? wr0 : 0) * K;
    const _Float16* Wrow1 = W + (size_t)(wr1 < N ? wr1 : 0) * K;

    v8f acc00 = {}, acc01 = {}, acc10 = {}, acc11 = {};
    for (int kb = 0; kb < K; kb += 32) {
        v16h a0 = loadA16h(Arow0, kb, half);
        v16h a1 = loadA16h(Arow1, kb, half);
        v16h b0 = loadB16h(Wrow0, kb, half);
        v16h b1 = loadB16h(Wrow1, kb, half);
        acc00 = __builtin_amdgcn_wmma_f32_16x16x32_f16(false, a0, false, b0, (short)0, acc00, false, false);
        acc01 = __builtin_amdgcn_wmma_f32_16x16x32_f16(false, a0, false, b1, (short)0, acc01, false, false);
        acc10 = __builtin_amdgcn_wmma_f32_16x16x32_f16(false, a1, false, b0, (short)0, acc10, false, false);
        acc11 = __builtin_amdgcn_wmma_f32_16x16x32_f16(false, a1, false, b1, (short)0, acc11, false, false);
    }

#pragma unroll
    for (int in = 0; in < 2; ++in) {
        int col = tn + 16 * in + l16;
        if (col < N) {
            float bv = bias ? bias[col] : 0.f;
#pragma unroll
            for (int im = 0; im < 2; ++im) {
                v8f acc = (im == 0) ? (in == 0 ? acc00 : acc01)
                                    : (in == 0 ? acc10 : acc11);
#pragma unroll
                for (int r = 0; r < 8; ++r) {
                    int row = tm + 16 * im + (half << 3) + r;   // lanes16-31 hold M+8
                    float v = acc[r] + bv;
                    if (act == 1)      v = v > 0.f ? v : 0.f;
                    else if (act == 2) v = v > 0.f ? v : 0.2f * v;
                    if (res) v += res[(size_t)row * N + col];
                    if (C)  C[(size_t)row * N + col]  = v;
                    if (Ch) Ch[(size_t)row * N + col] = (_Float16)v;
                }
            }
        }
    }
}

// ---------------------------------------------------------------------------
// FPS: one block per batch; per-thread 64 distances in registers; LDS argmax.
// ---------------------------------------------------------------------------
__global__ __launch_bounds__(1024)
void fps_kernel(const float* __restrict__ hits, float* __restrict__ centers)
{
    int b = blockIdx.x, tid = threadIdx.x;
    __shared__ float rd[1024];
    __shared__ int   ri[1024];
    const float* hb = hits + (size_t)b * NN * 5;
    float dist[64];
#pragma unroll
    for (int i = 0; i < 64; ++i) dist[i] = 1e10f;
    int far = 0;
    for (int it = 0; it < NPATCH; ++it) {
        float cx = hb[(size_t)far * 5 + 0];
        float cy = hb[(size_t)far * 5 + 1];
        float cz = hb[(size_t)far * 5 + 2];
        if (tid == 0) {
            centers[(b * NPATCH + it) * 3 + 0] = cx;
            centers[(b * NPATCH + it) * 3 + 1] = cy;
            centers[(b * NPATCH + it) * 3 + 2] = cz;
        }
        float bd = -1.f; int bi = 0x7fffffff;
#pragma unroll
        for (int i = 0; i < 64; ++i) {
            int n = tid + (i << 10);
            float dx = hb[(size_t)n * 5 + 0] - cx;
            float dy = hb[(size_t)n * 5 + 1] - cy;
            float dz = hb[(size_t)n * 5 + 2] - cz;
            float d  = fminf(dist[i], dx * dx + dy * dy + dz * dz);
            dist[i]  = d;
            if (d > bd || (d == bd && n < bi)) { bd = d; bi = n; }
        }
        rd[tid] = bd; ri[tid] = bi;
        __syncthreads();
        for (int s = 512; s > 0; s >>= 1) {
            if (tid < s) {
                float od = rd[tid + s]; int oi = ri[tid + s];
                if (od > rd[tid] || (od == rd[tid] && oi < ri[tid])) { rd[tid] = od; ri[tid] = oi; }
            }
            __syncthreads();
        }
        far = ri[0];
        __syncthreads();
    }
}

// ---------------------------------------------------------------------------
// kNN: block per (b,patch); per-thread top-32 candidates in LDS; 32 selection
// rounds reproduce top_k(ascending dist, ties -> smaller index).
// ---------------------------------------------------------------------------
__global__ __launch_bounds__(128)
void knn_kernel(const float* __restrict__ hits, const float* __restrict__ centers,
                int* __restrict__ gidx)
{
    int bg = blockIdx.x, b = bg >> 6, tid = threadIdx.x;
    __shared__ float cd[128 * 32];
    __shared__ int   ci[128 * 32];
    __shared__ float rd[128];
    __shared__ int   rp[128];
    float cx = centers[bg * 3 + 0], cy = centers[bg * 3 + 1], cz = centers[bg * 3 + 2];
    float* md = cd + tid * 32;
    int*   mi = ci + tid * 32;
    for (int s = 0; s < 32; ++s) { md[s] = 1e30f; mi[s] = 1 << 30; }
    float mx = 1e30f; int mslot = 0;
    const float* hb = hits + (size_t)b * NN * 5;
    for (int n = tid; n < NN; n += 128) {
        __builtin_prefetch(&hb[(size_t)(n + 128) * 5], 0, 0);
        float dx = hb[(size_t)n * 5 + 0] - cx;
        float dy = hb[(size_t)n * 5 + 1] - cy;
        float dz = hb[(size_t)n * 5 + 2] - cz;
        float d = dx * dx + dy * dy + dz * dz;
        if (d < mx) {
            md[mslot] = d; mi[mslot] = n;
            mx = -1.f;
            for (int s = 0; s < 32; ++s) {
                float v = md[s];
                if (v > mx) { mx = v; mslot = s; }
            }
        }
    }
    __syncthreads();
    for (int r = 0; r < KK; ++r) {
        float bd = 1e30f; int bslot = 0;
        for (int s = 0; s < 32; ++s) {
            float v = md[s];
            if (v < bd || (v == bd && mi[s] < mi[bslot])) { bd = v; bslot = s; }
        }
        rd[tid] = bd; rp[tid] = tid * 32 + bslot;
        __syncthreads();
        for (int s = 64; s > 0; s >>= 1) {
            if (tid < s) {
                float od = rd[tid + s]; int op = rp[tid + s];
                if (od < rd[tid] || (od == rd[tid] && ci[op] < ci[rp[tid]])) { rd[tid] = od; rp[tid] = op; }
            }
            __syncthreads();
        }
        if (tid == 0) { int p = rp[0]; gidx[bg * KK + r] = ci[p]; cd[p] = 1e30f; ci[p] = 1 << 30; }
        __syncthreads();
    }
}

// ---------------------------------------------------------------------------
// Mini-PointNet per patch: (K,5)->128, LN+relu, ->128, max over K
// ---------------------------------------------------------------------------
__global__ __launch_bounds__(128)
void pointnet_kernel(const float* __restrict__ hits, const int* __restrict__ gidx,
                     const float* __restrict__ centers,
                     const float* __restrict__ w1, const float* __restrict__ b1,
                     const float* __restrict__ lg, const float* __restrict__ lb,
                     const float* __restrict__ w2, const float* __restrict__ b2,
                     float* __restrict__ tokens)
{
    int bg = blockIdx.x, b = bg >> 6, j = threadIdx.x;
    __shared__ float pts[KK][5];
    __shared__ float t1[KK][ED];
    __shared__ float red[ED];
    float cx = centers[bg * 3 + 0], cy = centers[bg * 3 + 1], cz = centers[bg * 3 + 2];
    for (int e = j; e < KK * 5; e += 128) {
        int k = e / 5, c = e % 5;
        int idx = gidx[bg * KK + k];
        float v = hits[((size_t)b * NN + idx) * 5 + c];
        if (c == 0) v -= cx; else if (c == 1) v -= cy; else if (c == 2) v -= cz;
        pts[k][c] = v;
    }
    __syncthreads();
    for (int k = 0; k < KK; ++k) {
        float s = b1[j];
#pragma unroll
        for (int c = 0; c < 5; ++c) s += pts[k][c] * w1[j * 5 + c];
        t1[k][j] = s;
    }
    __syncthreads();
    for (int k = 0; k < KK; ++k) {
        float v = t1[k][j];
        red[j] = v; __syncthreads();
        for (int s = 64; s > 0; s >>= 1) { if (j < s) red[j] += red[j + s]; __syncthreads(); }
        float mean = red[0] * (1.f / ED); __syncthreads();
        float dv = v - mean;
        red[j] = dv * dv; __syncthreads();
        for (int s = 64; s > 0; s >>= 1) { if (j < s) red[j] += red[j + s]; __syncthreads(); }
        float var = red[0] * (1.f / ED); __syncthreads();
        float o = dv * rsqrtf(var + 1e-5f) * lg[j] + lb[j];
        t1[k][j] = fmaxf(o, 0.f);
        __syncthreads();
    }
    float best = -1e30f;
    for (int k = 0; k < KK; ++k) {
        float s = b2[j];
        for (int i = 0; i < ED; ++i) s += t1[k][i] * w2[j * ED + i];
        best = fmaxf(best, s);
    }
    tokens[(size_t)bg * ED + j] = best;
}

// ---------------------------------------------------------------------------
// Tiny MLP (3 -> hidden, relu, hidden -> hidden); used for center-embed & spos
// ---------------------------------------------------------------------------
__global__ void mlp2_kernel(const float* __restrict__ in3,
                            const float* __restrict__ w1, const float* __restrict__ b1,
                            const float* __restrict__ w2, const float* __restrict__ b2,
                            float* __restrict__ out, int hidden, int out_stride, int out_off)
{
    int m = blockIdx.x, j = threadIdx.x;
    __shared__ float h[128];
    float s = b1[j];
    for (int c = 0; c < 3; ++c) s += in3[m * 3 + c] * w1[j * 3 + c];
    h[j] = fmaxf(s, 0.f);
    __syncthreads();
    float o = b2[j];
    for (int i = 0; i < hidden; ++i) o += h[i] * w2[j * hidden + i];
    out[(size_t)m * out_stride + out_off + j] = o;
}

// ---------------------------------------------------------------------------
// LayerNorm (one row per block, blockDim == d), optional act; f16 output
// (LN outputs feed GEMMs only, which consume f16 fragments).
// ---------------------------------------------------------------------------
__global__ void layernorm_kernel(const float* __restrict__ x, const float* __restrict__ g,
                                 const float* __restrict__ b, _Float16* __restrict__ y,
                                 int d, int act)
{
    int row = blockIdx.x, j = threadIdx.x;
    __shared__ float red[128];
    float v = x[(size_t)row * d + j];
    red[j] = v; __syncthreads();
    for (int s = 64; s > 0; s >>= 1) { if (j < s && j + s < d) red[j] += red[j + s]; __syncthreads(); }
    float mean = red[0] / d; __syncthreads();
    float dv = v - mean;
    red[j] = dv * dv; __syncthreads();
    for (int s = 64; s > 0; s >>= 1) { if (j < s && j + s < d) red[j] += red[j + s]; __syncthreads(); }
    float var = red[0] / d;
    float o = dv * rsqrtf(var + 1e-5f) * g[j] + b[j];
    if (act == 1)      o = fmaxf(o, 0.f);
    else if (act == 2) o = o > 0.f ? o : 0.2f * o;
    y[(size_t)row * d + j] = (_Float16)o;
}

// ---------------------------------------------------------------------------
// Attention: block per (b,h), one thread per query; qkv rows are [q|k|v]
// Output f16 (feeds the projection GEMM only).
// ---------------------------------------------------------------------------
template<int S, int HD, int NH>
__global__ void attention_kernel(const float* __restrict__ qkv, _Float16* __restrict__ o)
{
    const int D = HD * NH;
    int bh = blockIdx.x, b = bh / NH, h = bh % NH;
    int q = threadIdx.x;
    const float* base = qkv + (size_t)(b * S) * (3 * D);
    float qv[HD];
#pragma unroll
    for (int j = 0; j < HD; ++j) qv[j] = base[(size_t)q * 3 * D + h * HD + j];
    float sc[S];
    float mx = -1e30f;
    float scale = rsqrtf((float)HD);
    for (int t = 0; t < S; ++t) {
        const float* kr = base + (size_t)t * 3 * D + D + h * HD;
        float s = 0.f;
#pragma unroll
        for (int j = 0; j < HD; ++j) s += qv[j] * kr[j];
        s *= scale;
        sc[t] = s; mx = fmaxf(mx, s);
    }
    float den = 0.f;
    for (int t = 0; t < S; ++t) { sc[t] = __expf(sc[t] - mx); den += sc[t]; }
    float inv = 1.f / den;
#pragma unroll
    for (int j = 0; j < HD; ++j) {
        float acc = 0.f;
        for (int t = 0; t < S; ++t) acc += sc[t] * base[(size_t)t * 3 * D + 2 * D + h * HD + j];
        o[(size_t)(b * S + q) * D + h * HD + j] = (_Float16)(acc * inv);
    }
}

// ---------------------------------------------------------------------------
// Small glue kernels
// ---------------------------------------------------------------------------
__global__ void argsort_kernel(const float* __restrict__ noise, int* __restrict__ shuf,
                               int* __restrict__ rest, float* __restrict__ mask_out)
{
    int b = blockIdx.x, i = threadIdx.x;
    __shared__ float nn[NPATCH];
    nn[i] = noise[b * NPATCH + i];
    __syncthreads();
    float ni = nn[i];
    int rank = 0;
    for (int j = 0; j < NPATCH; ++j)
        rank += (nn[j] < ni) || (nn[j] == ni && j < i);
    rest[b * NPATCH + i] = rank;
    shuf[b * NPATCH + rank] = i;
    mask_out[b * NPATCH + i] = (rank < NPATCH / 2) ? 0.f : 1.f;
}

__global__ void compose_x(const float* __restrict__ tok, const float* __restrict__ ce,
                          const float* __restrict__ pos, float* __restrict__ x)
{
    int m = blockIdx.x, j = threadIdx.x;
    x[(size_t)m * ED + j] = tok[(size_t)m * ED + j] + ce[(size_t)m * ED + j] + pos[(m & 63) * ED + j];
}

__global__ void gather_vis(const float* __restrict__ x, const int* __restrict__ shuf,
                           float* __restrict__ xv)
{
    int r = blockIdx.x, b = r >> 5, i = r & 31, j = threadIdx.x;
    int src = shuf[b * NPATCH + i];
    xv[(size_t)r * ED + j] = x[(size_t)(b * NPATCH + src) * ED + j];
}

__global__ void build_xfull(const float* __restrict__ xdec, const int* __restrict__ rest,
                            const float* __restrict__ mtok, float* __restrict__ xf)
{
    int m = blockIdx.x, b = m >> 6, j = threadIdx.x;   // 64 threads (DD)
    int ir = rest[m];
    xf[(size_t)m * DT + j] = (ir < 32) ? xdec[(size_t)(b * 32 + ir) * DD + j] : mtok[j];
}

// comb = concat(decoded, folding_seed, spatial_grid) zero-padded to 128 cols (f16)
__global__ void build_comb(const float* __restrict__ dec, const float* __restrict__ seed,
                           const float* __restrict__ gpts, _Float16* __restrict__ comb)
{
    int m = blockIdx.x, j = threadIdx.x;   // 128 threads
    float v;
    if (j < DT)            v = dec[(size_t)(m >> 5) * DT + j];
    else if (j < DT + 16)  v = seed[(m & 31) * 16 + (j - DT)];
    else if (j < DT + 19)  v = gpts[(m & 31) * 3 + (j - DT - 16)];
    else                   v = 0.f;
    comb[(size_t)m * 128 + j] = (_Float16)v;
}

// zero-pad a weight matrix [N,K] f32 -> [N,Kp] f16
__global__ void pad_w_kernel(const float* __restrict__ w, _Float16* __restrict__ wp, int K, int Kp)
{
    int n = blockIdx.x, k = threadIdx.x;   // Kp threads
    wp[(size_t)n * Kp + k] = (_Float16)((k < K) ? w[(size_t)n * K + k] : 0.f);
}

__global__ void f2h_kernel(const float* __restrict__ a, _Float16* __restrict__ o, int n)
{ int i = blockIdx.x * blockDim.x + threadIdx.x; if (i < n) o[i] = (_Float16)a[i]; }

__global__ void copy_f(const float* __restrict__ a, float* __restrict__ o, int n)
{ int i = blockIdx.x * blockDim.x + threadIdx.x; if (i < n) o[i] = a[i]; }

__global__ void i2f_kernel(const int* __restrict__ a, float* __restrict__ o, int n)
{ int i = blockIdx.x * blockDim.x + threadIdx.x; if (i < n) o[i] = (float)a[i]; }

// ---------------------------------------------------------------------------
// Host orchestration
// ---------------------------------------------------------------------------
static inline int gblocks(int M, int N) { int tiles = (M / 32) * ((N + 31) / 32); return (tiles + 7) / 8; }

extern "C" void kernel_launch(void* const* d_in, const int* in_sizes, int n_in,
                              void* d_out, int out_size, void* d_ws, size_t ws_size,
                              hipStream_t stream)
{
    const float* hits  = (const float*)d_in[0];
    const float* noise = (const float*)d_in[1];
    auto P = [&](int i) { return (const float*)d_in[i]; };

    float* out         = (float*)d_out;
    float* out_r       = out + OFF_R;
    float* out_mask    = out + OFF_MASK;
    float* out_decoded = out + OFF_DEC;
    float* out_centers = out + OFF_CENTERS;
    float* out_gidx    = out + OFF_GIDX;

    char* wp = (char*)d_ws;
    auto alloc = [&](size_t bytes) -> void* {
        void* r = (void*)wp; wp += (bytes + 255) & ~(size_t)255; return r;
    };
    int*      gidx   = (int*)     alloc((size_t)BB * NPATCH * KK * 4);
    int*      shuf   = (int*)     alloc((size_t)BB * NPATCH * 4);
    int*      rest   = (int*)     alloc((size_t)BB * NPATCH * 4);
    float*    tokens = (float*)   alloc((size_t)BB * NPATCH * ED * 4);
    float*    cebuf  = (float*)   alloc((size_t)BB * NPATCH * ED * 4);
    float*    xbuf   = (float*)   alloc((size_t)BB * NPATCH * ED * 4);
    float*    xv_a   = (float*)   alloc((size_t)BB * 32 * ED * 4);
    float*    xv_b   = (float*)   alloc((size_t)BB * 32 * ED * 4);
    _Float16* lnh    = (_Float16*)alloc((size_t)2048 * 128 * 2);
    float*    qkvbuf = (float*)   alloc((size_t)2048 * 384 * 4);
    _Float16* obufh  = (_Float16*)alloc((size_t)2048 * 128 * 2);
    _Float16* ffh    = (_Float16*)alloc((size_t)2048 * FFD * 2);
    _Float16* xvh    = (_Float16*)alloc((size_t)1024 * ED * 2);
    float*    xdec   = (float*)   alloc((size_t)1024 * DD * 4);
    float*    xf_a   = (float*)   alloc((size_t)2048 * DT * 4);
    float*    xf_b   = (float*)   alloc((size_t)2048 * DT * 4);
    _Float16* combh  = (_Float16*)alloc((size_t)65536 * 128 * 2);
    float*    r1     = (float*)   alloc((size_t)65536 * DT * 4);
    float*    r2     = (float*)   alloc((size_t)65536 * DT * 4);
    _Float16* r1h    = (_Float16*)alloc((size_t)65536 * DT * 2);
    _Float16* r2h    = (_Float16*)alloc((size_t)65536 * DT * 2);
    _Float16* whbuf  = (_Float16*)alloc((size_t)FFD * 128 * 2);   // weight staging (max 2048x128)

    auto convW = [&](const float* w, int n) {
        f2h_kernel<<<(n + 255) / 256, 256, 0, stream>>>(w, whbuf, n);
    };

    // --- grouping stage ---
    fps_kernel<<<BB, 1024, 0, stream>>>(hits, out_centers);
    knn_kernel<<<BB * NPATCH, 128, 0, stream>>>(hits, out_centers, gidx);
    pointnet_kernel<<<BB * NPATCH, 128, 0, stream>>>(hits, gidx, out_centers,
                                                     P(2), P(3), P(4), P(5), P(6), P(7), tokens);
    mlp2_kernel<<<BB * NPATCH, 128, 0, stream>>>(out_centers, P(8), P(9), P(10), P(11),
                                                 cebuf, 128, ED, 0);
    argsort_kernel<<<BB, NPATCH, 0, stream>>>(noise, shuf, rest, out_mask);
    compose_x<<<BB * NPATCH, ED, 0, stream>>>(tokens, cebuf, P(12), xbuf);
    gather_vis<<<BB * 32, ED, 0, stream>>>(xbuf, shuf, xv_a);

    // --- encoder: 6 layers, M=1024, d=128 ---
    for (int l = 0; l < 6; ++l) {
        int base = 13 + 12 * l;
        layernorm_kernel<<<1024, ED, 0, stream>>>(xv_a, P(base + 0), P(base + 1), lnh, ED, 0);
        convW(P(base + 2), 3 * ED * ED);
        gemm_wmma<<<gblocks(1024, 3 * ED), 256, 0, stream>>>(lnh, whbuf, P(base + 3),
                                                             nullptr, qkvbuf, nullptr, 1024, 3 * ED, ED, 0);
        attention_kernel<32, 16, 8><<<BB * 8, 32, 0, stream>>>(qkvbuf, obufh);
        convW(P(base + 4), ED * ED);
        gemm_wmma<<<gblocks(1024, ED), 256, 0, stream>>>(obufh, whbuf, P(base + 5),
                                                         xv_a, xv_b, nullptr, 1024, ED, ED, 0);
        layernorm_kernel<<<1024, ED, 0, stream>>>(xv_b, P(base + 6), P(base + 7), lnh, ED, 0);
        convW(P(base + 8), FFD * ED);
        gemm_wmma<<<gblocks(1024, FFD), 256, 0, stream>>>(lnh, whbuf, P(base + 9),
                                                          nullptr, nullptr, ffh, 1024, FFD, ED, 1);
        convW(P(base + 10), ED * FFD);
        gemm_wmma<<<gblocks(1024, ED), 256, 0, stream>>>(ffh, whbuf, P(base + 11),
                                                         xv_b, xv_a, nullptr, 1024, ED, FFD, 0);
    }

    // --- decoder input assembly ---
    f2h_kernel<<<(1024 * ED + 255) / 256, 256, 0, stream>>>(xv_a, xvh, 1024 * ED);
    convW(P(85), DD * ED);
    gemm_wmma<<<gblocks(1024, DD), 256, 0, stream>>>(xvh, whbuf, P(86), nullptr, xdec, nullptr,
                                                     1024, DD, ED, 0);
    build_xfull<<<BB * NPATCH, DD, 0, stream>>>(xdec, rest, P(87), xf_a);
    mlp2_kernel<<<BB * NPATCH, PD, 0, stream>>>(out_centers, P(88), P(89), P(90), P(91),
                                                xf_a, PD, DT, DD);

    // --- decoder: 4 layers, M=2048, d=96 ---
    for (int l = 0; l < 4; ++l) {
        int base = 92 + 12 * l;
        layernorm_kernel<<<2048, DT, 0, stream>>>(xf_a, P(base + 0), P(base + 1), lnh, DT, 0);
        convW(P(base + 2), 3 * DT * DT);
        gemm_wmma<<<gblocks(2048, 3 * DT), 256, 0, stream>>>(lnh, whbuf, P(base + 3),
                                                             nullptr, qkvbuf, nullptr, 2048, 3 * DT, DT, 0);
        attention_kernel<64, 24, 4><<<BB * 4, 64, 0, stream>>>(qkvbuf, obufh);
        convW(P(base + 4), DT * DT);
        gemm_wmma<<<gblocks(2048, DT), 256, 0, stream>>>(obufh, whbuf, P(base + 5),
                                                         xf_a, xf_b, nullptr, 2048, DT, DT, 0);
        layernorm_kernel<<<2048, DT, 0, stream>>>(xf_b, P(base + 6), P(base + 7), lnh, DT, 0);
        convW(P(base + 8), FFD * DT);
        gemm_wmma<<<gblocks(2048, FFD), 256, 0, stream>>>(lnh, whbuf, P(base + 9),
                                                          nullptr, nullptr, ffh, 2048, FFD, DT, 1);
        convW(P(base + 10), DT * FFD);
        gemm_wmma<<<gblocks(2048, DT), 256, 0, stream>>>(ffh, whbuf, P(base + 11),
                                                         xf_b, xf_a, nullptr, 2048, DT, FFD, 0);
    }

    // decoded output
    copy_f<<<(BB * NPATCH * DT + 255) / 256, 256, 0, stream>>>(xf_a, out_decoded, BB * NPATCH * DT);

    // --- reconstruction head over 65536 rows (K padded to 128) ---
    build_comb<<<65536, 128, 0, stream>>>(xf_a, P(140), P(141), combh);
    pad_w_kernel<<<DT, 128, 0, stream>>>(P(142), whbuf, DT + 19, 128);
    gemm_wmma<<<gblocks(65536, DT), 256, 0, stream>>>(combh, whbuf, P(143), nullptr, r1, nullptr,
                                                      65536, DT, 128, 0);
    layernorm_kernel<<<65536, DT, 0, stream>>>(r1, P(144), P(145), r1h, DT, 2);
    convW(P(146), DT * DT);
    gemm_wmma<<<gblocks(65536, DT), 256, 0, stream>>>(r1h, whbuf, P(147), nullptr, r2, nullptr,
                                                      65536, DT, DT, 0);
    layernorm_kernel<<<65536, DT, 0, stream>>>(r2, P(148), P(149), r2h, DT, 2);
    convW(P(150), OUTC * DT);
    gemm_wmma<<<gblocks(65536, OUTC), 256, 0, stream>>>(r2h, whbuf, P(151), nullptr, out_r, nullptr,
                                                        65536, OUTC, DT, 0);

    // group_idx output (as float values)
    i2f_kernel<<<(BB * NPATCH * KK + 255) / 256, 256, 0, stream>>>(gidx, out_gidx, BB * NPATCH * KK);
}
